// AdaptiveLabelPropagation_2929167695967
// MI455X (gfx1250) — compile-verified, matrix operand order checked by reading, not run on hardware
//
#include <hip/hip_runtime.h>
#include <math.h>

#define D_DIM 128
#define C_DIM 16
#define ALPHA_F 0.5f
#define EPS_LN 1e-5f
#define EPS_COS 1e-8f

typedef __attribute__((ext_vector_type(16))) _Float16 v16h;
typedef __attribute__((ext_vector_type(8)))  float    v8f;

// ---------------------------------------------------------------------------
// Kernel 1: h = features @ W^T + b  (N x 128) via WMMA f32 <- f16
// One block (8 waves) per 16-row slab; wave w computes output cols [16w,16w+16)
// A fragment layout (16-bit A 16x32, ISA 7.12.2): lane l holds M=l&15,
// half j -> K = (j<8?0:16) + 8*(l>>4) + (j&7). B assumed symmetric with N.
// D layout: VGPR r -> M = r + 8*(l>>4), N = l&15.
// ---------------------------------------------------------------------------
__global__ __launch_bounds__(256) void gemm_bias_kernel(
    const float* __restrict__ features, const float* __restrict__ W,
    const float* __restrict__ bias, float* __restrict__ h, int N)
{
    const int wave = threadIdx.x >> 5;
    const int lane = threadIdx.x & 31;
    const int mr   = lane & 15;
    const int khi  = (lane >> 4) << 3;   // 0 or 8
    const int row0 = blockIdx.x * 16;
    const int n0   = wave * 16;

    int arow = row0 + mr; if (arow >= N) arow = N - 1;  // clamp loads
    const float* __restrict__ Arow = features + (size_t)arow * D_DIM;
    const float* __restrict__ Brow = W + (size_t)(n0 + mr) * D_DIM;  // B[k][n]=W[n][k]

    v8f acc = {};
#pragma unroll
    for (int ks = 0; ks < 4; ++ks) {
        const int base = ks * 32;
        float4 a0 = *(const float4*)(Arow + base + khi);
        float4 a1 = *(const float4*)(Arow + base + khi + 4);
        float4 a2 = *(const float4*)(Arow + base + khi + 16);
        float4 a3 = *(const float4*)(Arow + base + khi + 20);
        float4 b0 = *(const float4*)(Brow + base + khi);
        float4 b1 = *(const float4*)(Brow + base + khi + 4);
        float4 b2 = *(const float4*)(Brow + base + khi + 16);
        float4 b3 = *(const float4*)(Brow + base + khi + 20);
        v16h av, bv;
        av[0]=(_Float16)a0.x; av[1]=(_Float16)a0.y; av[2]=(_Float16)a0.z; av[3]=(_Float16)a0.w;
        av[4]=(_Float16)a1.x; av[5]=(_Float16)a1.y; av[6]=(_Float16)a1.z; av[7]=(_Float16)a1.w;
        av[8]=(_Float16)a2.x; av[9]=(_Float16)a2.y; av[10]=(_Float16)a2.z; av[11]=(_Float16)a2.w;
        av[12]=(_Float16)a3.x; av[13]=(_Float16)a3.y; av[14]=(_Float16)a3.z; av[15]=(_Float16)a3.w;
        bv[0]=(_Float16)b0.x; bv[1]=(_Float16)b0.y; bv[2]=(_Float16)b0.z; bv[3]=(_Float16)b0.w;
        bv[4]=(_Float16)b1.x; bv[5]=(_Float16)b1.y; bv[6]=(_Float16)b1.z; bv[7]=(_Float16)b1.w;
        bv[8]=(_Float16)b2.x; bv[9]=(_Float16)b2.y; bv[10]=(_Float16)b2.z; bv[11]=(_Float16)b2.w;
        bv[12]=(_Float16)b3.x; bv[13]=(_Float16)b3.y; bv[14]=(_Float16)b3.z; bv[15]=(_Float16)b3.w;
        acc = __builtin_amdgcn_wmma_f32_16x16x32_f16(
            false, av, false, bv, (short)0, acc, false, false);
    }

    const float bn = bias[n0 + mr];
#pragma unroll
    for (int r = 0; r < 8; ++r) {
        const int m = row0 + khi + r;
        if (m < N) h[(size_t)m * D_DIM + n0 + mr] = acc[r] + bn;
    }
}

// ---------------------------------------------------------------------------
// Kernel 2: in-place per-row LayerNorm -> ReLU -> L2 normalize. Wave per row.
// ---------------------------------------------------------------------------
__global__ __launch_bounds__(256) void ln_relu_norm_kernel(
    float* __restrict__ h, const float* __restrict__ g,
    const float* __restrict__ be, int N)
{
    const int wave = threadIdx.x >> 5, lane = threadIdx.x & 31;
    const int row = blockIdx.x * 8 + wave;
    if (row >= N) return;
    float4 v = ((const float4*)(h + (size_t)row * D_DIM))[lane];

    float s = v.x + v.y + v.z + v.w;
#pragma unroll
    for (int m = 16; m >= 1; m >>= 1) s += __shfl_xor(s, m, 32);
    const float mu = s * (1.0f / 128.0f);

    float4 c = make_float4(v.x - mu, v.y - mu, v.z - mu, v.w - mu);
    float q = c.x*c.x + c.y*c.y + c.z*c.z + c.w*c.w;
#pragma unroll
    for (int m = 16; m >= 1; m >>= 1) q += __shfl_xor(q, m, 32);
    const float rs = rsqrtf(q * (1.0f / 128.0f) + EPS_LN);

    const float4 gv = ((const float4*)g)[lane];
    const float4 bv = ((const float4*)be)[lane];
    float4 y;
    y.x = fmaxf(c.x * rs * gv.x + bv.x, 0.0f);
    y.y = fmaxf(c.y * rs * gv.y + bv.y, 0.0f);
    y.z = fmaxf(c.z * rs * gv.z + bv.z, 0.0f);
    y.w = fmaxf(c.w * rs * gv.w + bv.w, 0.0f);

    float ss = y.x*y.x + y.y*y.y + y.z*y.z + y.w*y.w;
#pragma unroll
    for (int m = 16; m >= 1; m >>= 1) ss += __shfl_xor(ss, m, 32);
    const float inv = 1.0f / fmaxf(sqrtf(ss), EPS_COS);
    y.x *= inv; y.y *= inv; y.z *= inv; y.w *= inv;
    ((float4*)(h + (size_t)row * D_DIM))[lane] = y;
}

// ---------------------------------------------------------------------------
// Kernel 3: per-edge weight = sigmoid(ew) * dot(hn[src], hn[dst]). Wave/edge.
// ---------------------------------------------------------------------------
__global__ __launch_bounds__(256) void edge_weight_kernel(
    const float* __restrict__ hn, const int* __restrict__ src,
    const int* __restrict__ dst, const float* __restrict__ ew,
    float* __restrict__ wout, int E)
{
    const int wave = threadIdx.x >> 5, lane = threadIdx.x & 31;
    const long e = (long)blockIdx.x * 8 + wave;
    if (e >= E) return;
    const int s = src[e], d = dst[e];
    const float4 a = ((const float4*)(hn + (size_t)s * D_DIM))[lane];
    const float4 b = ((const float4*)(hn + (size_t)d * D_DIM))[lane];
    float p = a.x*b.x + a.y*b.y + a.z*b.z + a.w*b.w;
#pragma unroll
    for (int m = 16; m >= 1; m >>= 1) p += __shfl_xor(p, m, 32);
    if (lane == 0) {
        const float sg = 1.0f / (1.0f + __expf(-ew[0]));
        wout[e] = sg * p;
    }
}

// ---------------------------------------------------------------------------
// Kernel 4: zero next_logits (N*16) and total_w (N)
// ---------------------------------------------------------------------------
__global__ __launch_bounds__(256) void zero_kernel(
    float* __restrict__ nxt, float* __restrict__ tot, int N)
{
    const long t = (long)blockIdx.x * 256 + threadIdx.x;
    const long nc = (long)N * C_DIM;
    if (t < nc) nxt[t] = 0.0f;
    else if (t < nc + N) tot[t - nc] = 0.0f;
}

// ---------------------------------------------------------------------------
// Kernel 5: scatter-add over all 3 edge lists. Thread per (edge, class).
// ---------------------------------------------------------------------------
__global__ __launch_bounds__(256) void scatter_kernel(
    const int* __restrict__ s0, const int* __restrict__ d0,
    const int* __restrict__ s1, const int* __restrict__ d1,
    const int* __restrict__ s2, const int* __restrict__ d2,
    const float* __restrict__ wbuf, const float* __restrict__ cur,
    float* __restrict__ nxt, float* __restrict__ tot, int E)
{
    const long t = (long)blockIdx.x * 256 + threadIdx.x;
    const long ge = t >> 4;
    const int c = (int)(t & 15);
    if (ge >= 3L * E) return;
    long e = ge;
    const int* sp; const int* dp;
    if (e >= 2L * E)      { sp = s2; dp = d2; e -= 2L * E; }
    else if (e >= (long)E){ sp = s1; dp = d1; e -= E; }
    else                  { sp = s0; dp = d0; }
    const int s = sp[e], d = dp[e];
    const float w = wbuf[ge];
    unsafeAtomicAdd(nxt + (size_t)s * C_DIM + c, w * cur[(size_t)d * C_DIM + c]);
    if (c == 0) unsafeAtomicAdd(tot + s, w);
}

// ---------------------------------------------------------------------------
// Kernel 6: normalize by total_w (denom=1 when total_w<=0) and blend with init
// ---------------------------------------------------------------------------
__global__ __launch_bounds__(256) void norm_blend_kernel(
    const float* __restrict__ nxt, const float* __restrict__ tot,
    const float* __restrict__ init, float* __restrict__ out, int N)
{
    const long t = (long)blockIdx.x * 256 + threadIdx.x;
    if (t >= (long)N * C_DIM) return;
    const int n = (int)(t >> 4);
    const float tw = tot[n];
    const float denom = (tw > 0.0f) ? tw : 1.0f;
    const float v = nxt[t] / denom;
    out[t] = ALPHA_F * v + (1.0f - ALPHA_F) * init[t];
}

extern "C" void kernel_launch(void* const* d_in, const int* in_sizes, int n_in,
                              void* d_out, int out_size, void* d_ws, size_t ws_size,
                              hipStream_t stream) {
    const float* features = (const float*)d_in[0];
    const float* init     = (const float*)d_in[1];
    const float* W        = (const float*)d_in[2];
    const float* bias     = (const float*)d_in[3];
    const float* g        = (const float*)d_in[4];
    const float* be       = (const float*)d_in[5];
    const float* ew0      = (const float*)d_in[6];
    const float* ew1      = (const float*)d_in[7];
    const float* ew2      = (const float*)d_in[8];
    const int* s0 = (const int*)d_in[9],  *dd0 = (const int*)d_in[10];
    const int* s1 = (const int*)d_in[11], *dd1 = (const int*)d_in[12];
    const int* s2 = (const int*)d_in[13], *dd2 = (const int*)d_in[14];

    const int N = in_sizes[0] / D_DIM;
    const int E = in_sizes[9];

    float* hn   = (float*)d_ws;                 // N*128
    float* wbuf = hn + (size_t)N * D_DIM;       // 3*E
    float* cur  = wbuf + 3 * (size_t)E;         // N*16
    float* nxt  = cur + (size_t)N * C_DIM;      // N*16
    float* tot  = nxt + (size_t)N * C_DIM;      // N

    // Phase 1: feature transform (WMMA GEMM) + LN/ReLU/L2norm
    gemm_bias_kernel<<<(N + 15) / 16, 256, 0, stream>>>(features, W, bias, hn, N);
    ln_relu_norm_kernel<<<(N + 7) / 8, 256, 0, stream>>>(hn, g, be, N);

    // Phase 2: per-edge cosine weights (L2-resident gathers)
    const int eb = (E + 7) / 8;
    edge_weight_kernel<<<eb, 256, 0, stream>>>(hn, s0, dd0, ew0, wbuf,         E);
    edge_weight_kernel<<<eb, 256, 0, stream>>>(hn, s1, dd1, ew1, wbuf + E,     E);
    edge_weight_kernel<<<eb, 256, 0, stream>>>(hn, s2, dd2, ew2, wbuf + 2L*E,  E);

    // Phase 3: 5 propagation layers
    const long sthreads = 3L * E * C_DIM;
    const int  sblocks  = (int)((sthreads + 255) / 256);
    const int  zblocks  = (int)(((long)N * (C_DIM + 1) + 255) / 256);
    const int  nblocks  = (int)(((long)N * C_DIM + 255) / 256);

    const float* curp = init;
    for (int layer = 0; layer < 5; ++layer) {
        zero_kernel<<<zblocks, 256, 0, stream>>>(nxt, tot, N);
        scatter_kernel<<<sblocks, 256, 0, stream>>>(s0, dd0, s1, dd1, s2, dd2,
                                                    wbuf, curp, nxt, tot, E);
        float* outp = (layer == 4) ? (float*)d_out : cur;
        norm_blend_kernel<<<nblocks, 256, 0, stream>>>(nxt, tot, init, outp, N);
        curp = cur;
    }
}
